// GATGRUCellInversed_88639535055060
// MI455X (gfx1250) — matrix operand order repaired and checked by reading, not compiled, not run on hardware
//
#include <hip/hip_runtime.h>
#include <hip/hip_bf16.h>

typedef __bf16 bf16;
typedef __attribute__((ext_vector_type(16))) __bf16 v16bf;
typedef __attribute__((ext_vector_type(8)))  __bf16 v8bf;
typedef __attribute__((ext_vector_type(8)))  float  v8f;

#define NN     50000
#define NPREV  50000
#define NF     256
#define NHID   256
#define NHEADS 4
#define HD     64
#define DEG_IN 16
#define DEG_C  8
#define ALPHA  0.2f

// ---------------------------------------------------------------------------
// helpers
// ---------------------------------------------------------------------------
__device__ __forceinline__ void sched_fence() {
#if __has_builtin(__builtin_amdgcn_sched_barrier)
    __builtin_amdgcn_sched_barrier(0);   // nothing may be moved across this
#else
    asm volatile("" ::: "memory");
#endif
}

__device__ __forceinline__ float sigm_f(float x) {
    return 1.0f / (1.0f + __expf(-x));
}
__device__ __forceinline__ float tanh_f(float x) {
    return 2.0f / (1.0f + __expf(-2.0f * x)) - 1.0f;
}

// A fragment: 16x32 (MxK) bf16. lane L: row m=L&15; elems 0..7 -> k=k0+8*hi+i,
// elems 8..15 -> k=k0+16+8*hi+(i-8). arow must point at A + m*lda + hi*8.
__device__ __forceinline__ v16bf load_a_frag(const bf16* arow, int koff) {
    v8bf lo = *(const v8bf*)(arow + koff);
    v8bf hi = *(const v8bf*)(arow + koff + 16);
    v16bf r;
#pragma unroll
    for (int i = 0; i < 8; ++i) { r[i] = lo[i]; r[i + 8] = hi[i]; }
    return r;
}

// B fragment from col-major-by-output-column weights Bt[col][k] (ld = 256):
// lane L: col n=L&15; elems 0..7 -> k = k0 + 8*hi + i, elems 8..15 -> k = k0 + 16 + 8*hi + (i-8).
__device__ __forceinline__ v16bf load_b_frag(const bf16* Bt, int col, int k0,
                                             int lane) {
    int n  = lane & 15;
    int hi = lane >> 4;
    const bf16* p = Bt + (size_t)(col + n) * 256 + k0 + hi * 8;
    v8bf b0 = *(const v8bf*)(p);
    v8bf b1 = *(const v8bf*)(p + 16);
    v16bf r;
#pragma unroll
    for (int i = 0; i < 8; ++i) { r[i] = b0[i]; r[i + 8] = b1[i]; }
    return r;
}

// ---------------------------------------------------------------------------
// elementwise fp32 -> bf16
// ---------------------------------------------------------------------------
__global__ void __launch_bounds__(256) k_f2b(const float* __restrict__ in,
                                             bf16* __restrict__ out, int n) {
    int i = blockIdx.x * blockDim.x + threadIdx.x;
    if (i < n) out[i] = (bf16)in[i];
}

// W [NHEADS, 256, 64] -> Wt[c][k] bf16 where c = h*64+d  (65536 elems)
__global__ void __launch_bounds__(256) k_permW(const float* __restrict__ W,
                                               bf16* __restrict__ Wt) {
    int i = blockIdx.x * blockDim.x + threadIdx.x;   // i = c*256 + k
    if (i >= 256 * 256) return;
    int c = i >> 8, k = i & 255;
    Wt[i] = (bf16)W[((size_t)(c >> 6) * 256 + k) * 64 + (c & 63)];
}

// ---------------------------------------------------------------------------
// GEMM: C[M,256] = A[M,256] (bf16) x Bt (col-major-by-col, bf16) [+ bias]
// grid = M/16 blocks, 32 threads (1 wave). A block read once; 16 N-tiles.
// B fragments double-buffered (depth 2) so WMMAs overlap B-loads.
// ---------------------------------------------------------------------------
template <bool HASBIAS>
__global__ void __launch_bounds__(32, 1)
k_gemm_bf16(const bf16* __restrict__ A, const bf16* __restrict__ Bt,
            const float* __restrict__ bias, bf16* __restrict__ C) {
    const int lane = threadIdx.x;
    const int R    = blockIdx.x * 16;
    const int m    = lane & 15;
    const int hi   = lane >> 4;

    const bf16* arow = A + (size_t)(R + m) * 256 + hi * 8;
    v16bf a[8];
#pragma unroll
    for (int kc = 0; kc < 8; ++kc) a[kc] = load_a_frag(arow, kc * 32);

    for (int t = 0; t < 16; ++t) {
        v8f acc = {};
        v16bf b0 = load_b_frag(Bt, t * 16, 0, lane);
        v16bf b1 = load_b_frag(Bt, t * 16, 32, lane);
#pragma unroll
        for (int kc = 0; kc < 8; ++kc) {
            v16bf bn = b0;
            if (kc < 6) bn = load_b_frag(Bt, t * 16, (kc + 2) * 32, lane);
            acc = __builtin_amdgcn_wmma_f32_16x16x32_bf16(
                false, a[kc], false, b0, (short)0, acc, false, false);
            b0 = b1;
            b1 = bn;
        }
        int   c  = t * 16 + m;                 // this lane's output column
        float bv = HASBIAS ? bias[c] : 0.0f;
#pragma unroll
        for (int j = 0; j < 8; ++j) {
            int row = R + j + 8 * hi;
            C[(size_t)row * 256 + c] = (bf16)(acc[j] + bv);
        }
    }
}

// ---------------------------------------------------------------------------
// attention projections: out[n*4+h] = ab[h] + sum_d T[n,h*64+d]*a[h*128+off+d]
// ---------------------------------------------------------------------------
__global__ void __launch_bounds__(256)
k_att_proj(const bf16* __restrict__ T, const float* __restrict__ avec,
           const float* __restrict__ ab, float* __restrict__ out, int nnodes,
           int a_off) {
    int i = blockIdx.x * blockDim.x + threadIdx.x;
    if (i >= nnodes * NHEADS) return;
    int n = i >> 2, h = i & 3;
    const bf16*  row = T + (size_t)n * 256 + h * 64;
    const float* a   = avec + h * 128 + a_off;
    float s = ab ? ab[h] : 0.0f;
#pragma unroll 8
    for (int d = 0; d < 64; ++d) s += (float)row[d] * a[d];
    out[i] = s;
}

// ---------------------------------------------------------------------------
// intra-turn GAT aggregation: 1 block (256 thr) per dst node, 16 edges
// edges for dst n are contiguous: [n*16, n*16+16)
// ---------------------------------------------------------------------------
__global__ void __launch_bounds__(256)
k_gat_intra(const bf16* __restrict__ Wh, const float* __restrict__ es,
            const float* __restrict__ ed, const int* __restrict__ src,
            bf16* __restrict__ xout) {
    __shared__ int   s_src[DEG_IN];
    __shared__ float s_l[NHEADS][DEG_IN];
    int n = blockIdx.x, t = threadIdx.x;

    if (t < DEG_IN) s_src[t] = src[n * DEG_IN + t];
    __syncthreads();
    if (t < NHEADS * DEG_IN) {
        int h = t >> 4, e = t & 15;
        float v = es[s_src[e] * 4 + h] + ed[n * 4 + h];
        s_l[h][e] = (v >= 0.0f) ? v : ALPHA * v;
    }
    __syncthreads();
    if (t < NHEADS) {
        float mx = -1e30f;
#pragma unroll
        for (int e = 0; e < DEG_IN; ++e) mx = fmaxf(mx, s_l[t][e]);
        float sum = 0.0f;
#pragma unroll
        for (int e = 0; e < DEG_IN; ++e) {
            float ex = __expf(s_l[t][e] - mx);
            s_l[t][e] = ex;
            sum += ex;
        }
        float inv = 1.0f / fmaxf(sum, 1e-9f);
#pragma unroll
        for (int e = 0; e < DEG_IN; ++e) s_l[t][e] *= inv;
    }
    __syncthreads();
    int   h   = t >> 6;
    float acc = 0.0f;
#pragma unroll
    for (int e = 0; e < DEG_IN; ++e)
        acc += s_l[h][e] * (float)Wh[(size_t)s_src[e] * 256 + t];
    xout[(size_t)n * 256 + t] = (bf16)acc;
}

// ---------------------------------------------------------------------------
// cross-turn GAT: counter + support edge sets (8 each), h = 0.5*(hc+hs)
// ---------------------------------------------------------------------------
__global__ void __launch_bounds__(256)
k_gat_cross(const bf16* __restrict__ Whp, const float* __restrict__ esp,
            const float* __restrict__ edx, const int* __restrict__ srcC,
            const int* __restrict__ srcS, bf16* __restrict__ hout) {
    __shared__ int   sc[DEG_C], ss[DEG_C];
    __shared__ float lc[NHEADS][DEG_C], ls[NHEADS][DEG_C];
    int n = blockIdx.x, t = threadIdx.x;

    if (t < DEG_C)            sc[t]     = srcC[n * DEG_C + t];
    else if (t < 2 * DEG_C)   ss[t - 8] = srcS[n * DEG_C + (t - 8)];
    __syncthreads();
    if (t < NHEADS * 16) {
        int h = t >> 4, e = t & 15;
        float base = edx[n * 4 + h];
        if (e < DEG_C) {
            float v = esp[sc[e] * 4 + h] + base;
            lc[h][e] = (v >= 0.0f) ? v : ALPHA * v;
        } else {
            float v = esp[ss[e - 8] * 4 + h] + base;
            ls[h][e - 8] = (v >= 0.0f) ? v : ALPHA * v;
        }
    }
    __syncthreads();
    if (t < 2 * NHEADS) {
        float* row = (t < NHEADS) ? lc[t] : ls[t - NHEADS];
        float mx = -1e30f;
#pragma unroll
        for (int e = 0; e < DEG_C; ++e) mx = fmaxf(mx, row[e]);
        float sum = 0.0f;
#pragma unroll
        for (int e = 0; e < DEG_C; ++e) {
            float ex = __expf(row[e] - mx);
            row[e] = ex;
            sum += ex;
        }
        float inv = 1.0f / fmaxf(sum, 1e-9f);
#pragma unroll
        for (int e = 0; e < DEG_C; ++e) row[e] *= inv;
    }
    __syncthreads();
    int   h   = t >> 6;
    float acc = 0.0f;
#pragma unroll
    for (int e = 0; e < DEG_C; ++e)
        acc += lc[h][e] * (float)Whp[(size_t)sc[e] * 256 + t];
#pragma unroll
    for (int e = 0; e < DEG_C; ++e)
        acc += ls[h][e] * (float)Whp[(size_t)ss[e] * 256 + t];
    hout[(size_t)n * 256 + t] = (bf16)(0.5f * acc);
}

// ---------------------------------------------------------------------------
// fused GRUCell
// ---------------------------------------------------------------------------
// accumulate one (ih, hh) WMMA tile pair at weight column `col`, with depth-1
// prefetch of the next k-step's B fragments so WMMAs overlap B-loads.
__device__ __forceinline__ void gru_accum_pair(const bf16* __restrict__ Wih,
                                               const bf16* __restrict__ Whh,
                                               int col, int lane,
                                               const v16bf (&ax)[8],
                                               const v16bf (&ah)[8],
                                               v8f& gi, v8f& gh) {
    v16bf bi = load_b_frag(Wih, col, 0, lane);
    v16bf bh = load_b_frag(Whh, col, 0, lane);
#pragma unroll
    for (int kc = 0; kc < 8; ++kc) {
        v16bf bi_n = bi, bh_n = bh;
        if (kc < 7) {
            bi_n = load_b_frag(Wih, col, (kc + 1) * 32, lane);
            bh_n = load_b_frag(Whh, col, (kc + 1) * 32, lane);
        }
        gi = __builtin_amdgcn_wmma_f32_16x16x32_bf16(
            false, ax[kc], false, bi, (short)0, gi, false, false);
        gh = __builtin_amdgcn_wmma_f32_16x16x32_bf16(
            false, ah[kc], false, bh, (short)0, gh, false, false);
        bi = bi_n;
        bh = bh_n;
    }
}

// per 16-row tile: preload x & h A-blocks, then for each 16-col output tile
// compute gate tiles in 3 low-pressure passes (r, z, n) separated by
// scheduling fences (prevents pass fusion -> A-fragment spills), apply GRU
// math, write fp32 output. Wih/Whh are bf16 [768 cols][256 k] (torch layout).
__global__ void __launch_bounds__(32, 1)
k_gru_fused(const bf16* __restrict__ Xb, const bf16* __restrict__ Hb,
            const bf16* __restrict__ Wih, const bf16* __restrict__ Whh,
            const float* __restrict__ b_ih, const float* __restrict__ b_hh,
            float* __restrict__ out) {
    const int lane = threadIdx.x;
    const int R    = blockIdx.x * 16;
    const int m    = lane & 15;
    const int hi   = lane >> 4;

    const bf16* xrow = Xb + (size_t)(R + m) * 256 + hi * 8;
    const bf16* hrow = Hb + (size_t)(R + m) * 256 + hi * 8;
    v16bf ax[8], ah[8];
#pragma unroll
    for (int kc = 0; kc < 8; ++kc) {
        ax[kc] = load_a_frag(xrow, kc * 32);
        ah[kc] = load_a_frag(hrow, kc * 32);
    }

    for (int t = 0; t < 16; ++t) {
        const int c = t * 16 + m;
        float rv[8], zv[8];

        {   // pass 1: reset gate r
            v8f gi = {}, gh = {};
            gru_accum_pair(Wih, Whh, t * 16, lane, ax, ah, gi, gh);
            float bi = b_ih[c], bh = b_hh[c];
#pragma unroll
            for (int j = 0; j < 8; ++j)
                rv[j] = sigm_f(gi[j] + bi + gh[j] + bh);
        }
        sched_fence();
        {   // pass 2: update gate z
            v8f gi = {}, gh = {};
            gru_accum_pair(Wih, Whh, 256 + t * 16, lane, ax, ah, gi, gh);
            float bi = b_ih[256 + c], bh = b_hh[256 + c];
#pragma unroll
            for (int j = 0; j < 8; ++j)
                zv[j] = sigm_f(gi[j] + bi + gh[j] + bh);
        }
        sched_fence();
        {   // pass 3: candidate n + output
            v8f gi = {}, gh = {};
            gru_accum_pair(Wih, Whh, 512 + t * 16, lane, ax, ah, gi, gh);
            float bi = b_ih[512 + c], bh = b_hh[512 + c];
#pragma unroll
            for (int j = 0; j < 8; ++j) {
                int   row = R + j + 8 * hi;
                float nn  = tanh_f(gi[j] + bi + rv[j] * (gh[j] + bh));
                float hv  = (float)Hb[(size_t)row * 256 + c];
                out[(size_t)row * 256 + c] =
                    (1.0f - zv[j]) * nn + zv[j] * hv;
            }
        }
        sched_fence();
    }
}

// ---------------------------------------------------------------------------
// launch
// ---------------------------------------------------------------------------
extern "C" void kernel_launch(void* const* d_in, const int* in_sizes, int n_in,
                              void* d_out, int out_size, void* d_ws,
                              size_t ws_size, hipStream_t stream) {
    const float* h_t      = (const float*)d_in[0];
    const float* h_prev   = (const float*)d_in[1];
    const float* W_intra  = (const float*)d_in[2];
    const float* b_intra  = (const float*)d_in[3];
    const float* a_intra  = (const float*)d_in[4];
    const float* ab_intra = (const float*)d_in[5];
    const float* W_cross  = (const float*)d_in[6];
    const float* a_cross  = (const float*)d_in[7];
    const float* W_ih     = (const float*)d_in[8];
    const float* W_hh     = (const float*)d_in[9];
    const float* b_ih     = (const float*)d_in[10];
    const float* b_hh     = (const float*)d_in[11];
    const int*   src_in   = (const int*)d_in[12];
    const int*   src_c    = (const int*)d_in[14];
    const int*   src_s    = (const int*)d_in[16];
    float*       out      = (float*)d_out;

    // workspace carve (256B aligned)
    char* p     = (char*)d_ws;
    auto  alloc = [&](size_t bytes) {
        void* r = (void*)p;
        p += (bytes + 255) & ~(size_t)255;
        return r;
    };
    const size_t NE = (size_t)NN * 256;           // 12.8M elements

    bf16*  ht_bf    = (bf16*)alloc(NE * 2);
    bf16*  hprev_bf = (bf16*)alloc(NE * 2);
    bf16*  Wh_bf    = (bf16*)alloc(NE * 2);
    bf16*  x_bf     = (bf16*)alloc(NE * 2);
    bf16*  Whp_bf   = (bf16*)alloc(NE * 2);
    bf16*  Whx_bf   = (bf16*)alloc(NE * 2);
    bf16*  h_bf     = (bf16*)alloc(NE * 2);
    bf16*  Wi_t     = (bf16*)alloc(256 * 256 * 2);
    bf16*  Wc_t     = (bf16*)alloc(256 * 256 * 2);
    bf16*  Wih_bf   = (bf16*)alloc(768 * 256 * 2);
    bf16*  Whh_bf   = (bf16*)alloc(768 * 256 * 2);
    float* es       = (float*)alloc((size_t)NN * 4 * 4);
    float* ed       = (float*)alloc((size_t)NN * 4 * 4);
    float* es_p     = (float*)alloc((size_t)NPREV * 4 * 4);
    float* ed_x     = (float*)alloc((size_t)NN * 4 * 4);

    const int TB     = 256;
    const int mtiles = NN / 16;                   // 3125

    // 1) conversions / weight prep
    k_f2b<<<(int)((NE + TB - 1) / TB), TB, 0, stream>>>(h_t, ht_bf, (int)NE);
    k_f2b<<<(int)((NE + TB - 1) / TB), TB, 0, stream>>>(h_prev, hprev_bf,
                                                        (int)NE);
    k_permW<<<256, TB, 0, stream>>>(W_intra, Wi_t);
    k_permW<<<256, TB, 0, stream>>>(W_cross, Wc_t);
    k_f2b<<<768, TB, 0, stream>>>(W_ih, Wih_bf, 768 * 256);
    k_f2b<<<768, TB, 0, stream>>>(W_hh, Whh_bf, 768 * 256);

    // 2) intra: Wh = h_t @ W_intra + b
    k_gemm_bf16<true><<<mtiles, 32, 0, stream>>>(ht_bf, Wi_t, b_intra, Wh_bf);
    int napj = (NN * 4 + TB - 1) / TB;
    k_att_proj<<<napj, TB, 0, stream>>>(Wh_bf, a_intra, nullptr, es, NN, 0);
    k_att_proj<<<napj, TB, 0, stream>>>(Wh_bf, a_intra, ab_intra, ed, NN, 64);
    k_gat_intra<<<NN, TB, 0, stream>>>(Wh_bf, es, ed, src_in, x_bf);

    // 3) cross: Whp = h_prev @ W_cross ; Whx = x @ W_cross (shared weights)
    k_gemm_bf16<false><<<NPREV / 16, 32, 0, stream>>>(hprev_bf, Wc_t, nullptr,
                                                      Whp_bf);
    k_gemm_bf16<false><<<mtiles, 32, 0, stream>>>(x_bf, Wc_t, nullptr, Whx_bf);
    k_att_proj<<<napj, TB, 0, stream>>>(Whp_bf, a_cross, nullptr, es_p, NPREV,
                                        0);
    k_att_proj<<<napj, TB, 0, stream>>>(Whx_bf, a_cross, nullptr, ed_x, NN, 64);
    k_gat_cross<<<NN, TB, 0, stream>>>(Whp_bf, es_p, ed_x, src_c, src_s, h_bf);

    // 4) fused GRUCell
    k_gru_fused<<<mtiles, 32, 0, stream>>>(x_bf, h_bf, Wih_bf, Whh_bf, b_ih,
                                           b_hh, out);
}
